// HybridFCAttention_14061722927950
// MI455X (gfx1250) — matrix operand order repaired and checked by reading, not compiled, not run on hardware
//
#include <hip/hip_runtime.h>

typedef unsigned short u16;
typedef __attribute__((ext_vector_type(16))) __bf16 v16bf;
typedef __attribute__((ext_vector_type(8)))  float   v8f;
typedef __attribute__((ext_vector_type(4)))  float   v4f;

#define DEV __device__ __forceinline__

constexpr int Bc = 2, Sc = 2048, Dc = 1024, Hc = 16, HDc = 64;
constexpr int Mrows = Bc * Sc;  // 4096

union Frag { v16bf v; v4f f[2]; };

DEV u16 f2bf(float x) {
  unsigned u = __float_as_uint(x);
  u += 0x7fffu + ((u >> 16) & 1u);  // round-to-nearest-even
  return (u16)(u >> 16);
}

DEV v4f ldg4(const u16* p) { return *(const v4f*)p; }

DEV v8f wmma_bf16(v16bf a, v16bf b, v8f c) {
  // D = A(16x32 bf16) * B(32x16 bf16) + C(16x16 f32)
  return __builtin_amdgcn_wmma_f32_16x16x32_bf16(
      /*neg_a=*/false, a, /*neg_b=*/false, b,
      /*c_mod=*/(short)0, c, /*reuse_a=*/false, /*reuse_b=*/false);
}

// ---- CDNA5 async Global->LDS copy (ASYNCcnt-tracked), per ISA 15.18.3 ----
DEV void async_b128(unsigned lds_off, const u16* gptr) {
  asm volatile("global_load_async_to_lds_b128 %0, %1, off"
               :: "v"(lds_off), "v"((unsigned long long)(size_t)gptr)
               : "memory");
}
DEV void wait_async_le8() { asm volatile("s_wait_asynccnt 0x8" ::: "memory"); }
DEV void wait_async_0()   { asm volatile("s_wait_asynccnt 0x0" ::: "memory"); }

// Flat/generic pointer to a __shared__ object: low 32 bits are the LDS offset.
DEV unsigned lds_off_of(const void* p) { return (unsigned)(size_t)p; }

// ---------------------------------------------------------------------------
// fp32 -> bf16 conversion (grid-stride)
// ---------------------------------------------------------------------------
__global__ __launch_bounds__(256) void cvt_bf16_k(const float* __restrict__ in,
                                                  u16* __restrict__ out, int n) {
  int i = blockIdx.x * blockDim.x + threadIdx.x;
  int stride = gridDim.x * blockDim.x;
  for (; i < n; i += stride) out[i] = f2bf(in[i]);
}

// W[k*D+n] (fp32, KxN row-major) -> Wt[n*D+k] (bf16, N-major)
__global__ __launch_bounds__(256) void cvt_wT_k(const float* __restrict__ W,
                                                u16* __restrict__ Wt) {
  int idx = blockIdx.x * blockDim.x + threadIdx.x;  // idx = n*D + k
  int n = idx >> 10, k = idx & (Dc - 1);
  Wt[idx] = f2bf(W[(size_t)k * Dc + n]);
}

// ---------------------------------------------------------------------------
// Tiled bf16 WMMA GEMM: C[4096,1024] = A[4096,1024] * Bt[1024,1024]^T + bias
// Block 256 thr = 8 waves (4x2); wave tile 32x64; block tile 128x128.
// kTile = 64: A/B tiles (128x64 bf16, 16 KB each) double-buffered in LDS via
// global_load_async_to_lds_b128 (8 async ops/lane/stage, ASYNCcnt pipelined).
// EPI 0: RoPE + store bf16 [B,H,S,HD]                  (Q / K projections)
// EPI 1: store bf16 transposed [B,H,HD,S]              (V projection)
// EPI 2: store fp32 row-major [B,S,D]                  (output projection)
// ---------------------------------------------------------------------------
enum { EPI_ROPE = 0, EPI_V = 1, EPI_OUT = 2 };
constexpr int KT = 64;  // k-tile depth

template <int EPI>
__global__ __launch_bounds__(256)
void gemm_wmma_k(const u16* __restrict__ A, const u16* __restrict__ Bt,
                 const float* __restrict__ bias, void* __restrict__ dstv) {
  __shared__ u16 sA[2][128 * KT];  // 16 KB per buffer
  __shared__ u16 sB[2][128 * KT];  // 16 KB per buffer

  const int tid = threadIdx.x;
  const int lane = tid & 31;
  const int wave = tid >> 5;
  const int ln = lane & 15, half = lane >> 4;
  const int mw = wave >> 1, nw = wave & 1;
  const int rowBlock = blockIdx.y * 128;
  const int colBlock = blockIdx.x * 128;
  const int rowBase = mw * 32;   // wave tile origin within block (rows)
  const int colBase = nw * 64;   // wave tile origin within block (cols)
  const int kh = half * 8;       // A: K sub-offset per lane half
  const int khB = half * 8;      // B: K sub-offset per lane half

  const unsigned sAo = lds_off_of(&sA[0][0]);
  const unsigned sBo = lds_off_of(&sB[0][0]);

  // Stage one 128x64 A tile + 128x64 B tile into LDS buffer `buf`:
  // 1024 + 1024 b128 transfers, 256 threads -> 4 + 4 async ops per lane.
  auto stage = [&](int k0, int buf) {
#pragma unroll
    for (int t = 0; t < 4; ++t) {
      int i = tid + t * 256;                 // 0..1023
      int row = i >> 3, c = (i & 7) * 8;     // 8 bf16 per b128
      unsigned lo = (unsigned)(buf * (128 * KT) + row * KT + c) * 2u;
      async_b128(sAo + lo, A + (size_t)(rowBlock + row) * Dc + k0 + c);
      async_b128(sBo + lo, Bt + (size_t)(colBlock + row) * Dc + k0 + c);
    }
  };

  v8f c[2][4] = {};

  stage(0, 0);
  for (int kt = 0; kt < Dc / KT; ++kt) {
    const int buf = kt & 1;
    if (kt + 1 < Dc / KT) {
      stage((kt + 1) * KT, buf ^ 1);  // prefetch next tile (ASYNCcnt += 8)
      wait_async_le8();               // in-order: current tile resident
    } else {
      wait_async_0();
    }
    __syncthreads();  // all waves' copies for `buf` complete

    const u16* aBuf = &sA[buf][0];
    const u16* bBuf = &sB[buf][0];
#pragma unroll
    for (int ks = 0; ks < 2; ++ks) {  // two 32-wide sub-steps -> 16 WMMAs
      Frag a[2], b[4];
#pragma unroll
      for (int mi = 0; mi < 2; ++mi) {
        const u16* ap = aBuf + (rowBase + mi * 16 + ln) * KT + ks * 32;
        a[mi].f[0] = *(const v4f*)(ap + kh);
        a[mi].f[1] = *(const v4f*)(ap + kh + 16);
      }
#pragma unroll
      for (int ni = 0; ni < 4; ++ni) {
        const u16* bp = bBuf + (colBase + ni * 16 + ln) * KT + ks * 32 + khB;
        b[ni].f[0] = *(const v4f*)(bp);
        b[ni].f[1] = *(const v4f*)(bp + 16);
      }
#pragma unroll
      for (int mi = 0; mi < 2; ++mi)
#pragma unroll
        for (int ni = 0; ni < 4; ++ni)
          c[mi][ni] = wmma_bf16(a[mi].v, b[ni].v, c[mi][ni]);
    }

    __syncthreads();  // reads done before `buf` is overwritten next iter
  }

#pragma unroll
  for (int ni = 0; ni < 4; ++ni) {
    float bc = bias[colBlock + colBase + ni * 16 + ln];
#pragma unroll
    for (int mi = 0; mi < 2; ++mi)
#pragma unroll
      for (int r = 0; r < 8; ++r) c[mi][ni][r] += bc;
  }

  const int gRow = rowBlock + rowBase;
  const int gCol = colBlock + colBase;

  if constexpr (EPI == EPI_OUT) {
    float* out = (float*)dstv;
#pragma unroll
    for (int mi = 0; mi < 2; ++mi)
#pragma unroll
      for (int ni = 0; ni < 4; ++ni)
#pragma unroll
        for (int r = 0; r < 8; ++r) {
          int row = gRow + mi * 16 + r + 8 * half;
          int col = gCol + ni * 16 + ln;
          out[(size_t)row * Dc + col] = c[mi][ni][r];
        }
  } else if constexpr (EPI == EPI_V) {
    u16* dst = (u16*)dstv;
#pragma unroll
    for (int mi = 0; mi < 2; ++mi)
#pragma unroll
      for (int ni = 0; ni < 4; ++ni)
#pragma unroll
        for (int r = 0; r < 8; ++r) {
          int row = gRow + mi * 16 + r + 8 * half;
          int bb = row >> 11, s = row & (Sc - 1);
          int col = gCol + ni * 16 + ln;
          int h = col >> 6, d = col & 63;
          dst[((size_t)(bb * Hc + h) * HDc + d) * Sc + s] = f2bf(c[mi][ni][r]);
        }
  } else {  // EPI_ROPE: wave owns a whole 64-wide head -> pairing in regs
    u16* dst = (u16*)dstv;
    const int h = gCol >> 6;
    const float kfreq = -0.28782313662425572f;  // -ln(10000)/32
    const float invA = __expf(kfreq * (float)ln);
    const float invB = __expf(kfreq * (float)(16 + ln));
#pragma unroll
    for (int mi = 0; mi < 2; ++mi) {
#pragma unroll
      for (int r = 0; r < 8; ++r) {
        int row = gRow + mi * 16 + r + 8 * half;
        int bb = row >> 11;
        int s = row & (Sc - 1);
        float pos = (float)s;
#pragma unroll
        for (int nf = 0; nf < 4; ++nf) {
          int d = nf * 16 + ln;
          float th = pos * ((nf & 1) ? invB : invA);
          float sn, cs;
          __sincosf(th, &sn, &cs);
          float x = c[mi][nf][r];
          float y = c[mi][nf ^ 2][r];  // partner feature d +/- 32
          float val = (d < 32) ? (x * cs - y * sn) : (x * cs + y * sn);
          dst[((size_t)(bb * Hc + h) * Sc + s) * HDc + d] = f2bf(val);
        }
      }
    }
  }
}

// ---------------------------------------------------------------------------
// Flash attention with analytic FC mask.
// Grid: (B*H, S/64); block 128 thr = 4 waves; wave owns 16 q-rows.
// Q,K: bf16 [B,H,S,HD] (RoPE'd). Vt: bf16 [B,H,HD,S]. O: bf16 [B,S,D].
// ---------------------------------------------------------------------------
__global__ __launch_bounds__(128)
void attn_k(const u16* __restrict__ Q, const u16* __restrict__ K,
            const u16* __restrict__ Vt, const int* __restrict__ bar,
            const int* __restrict__ inst, const int* __restrict__ layer,
            u16* __restrict__ O) {
  const int bh = blockIdx.x;
  const int b = bh >> 4, h = bh & 15;
  const int qb = blockIdx.y;
  const int lane = threadIdx.x & 31, wave = threadIdx.x >> 5;
  const int ln = lane & 15, half = lane >> 4;
  const int kh = half * 8, khB = half * 8;
  const int qRowBase = qb * 64 + wave * 16;

  __shared__ u16 lds[4 * 16 * 64];
  u16* pl = lds + wave * (16 * 64);  // per-wave private P tile 16x64

  // Q A-fragments (16 rows x 64 HD), half h holds HD 32kc+[8h,8h+8) and 32kc+[16+8h,24+8h)
  Frag qa[2];
  {
    const u16* qp = Q + ((size_t)bh * Sc + qRowBase + ln) * HDc;
#pragma unroll
    for (int kc = 0; kc < 2; ++kc) {
      qa[kc].f[0] = ldg4(qp + kc * 32 + kh);
      qa[kc].f[1] = ldg4(qp + kc * 32 + kh + 16);
    }
  }
  int barq[8], instq[8];
#pragma unroll
  for (int r = 0; r < 8; ++r) {
    int row = qRowBase + r + 8 * half;
    barq[r] = bar[b * Sc + row];
    instq[r] = inst[b * Sc + row];
  }
  const bool fc = (layer[0] < 4);

  float mrow[8], lrow[8];
#pragma unroll
  for (int r = 0; r < 8; ++r) { mrow[r] = -3.0e38f; lrow[r] = 0.0f; }
  v8f o[4] = {};

  const int nkb = qb + 1;  // causal: key blocks up to and including diagonal
  for (int kb = 0; kb < nkb; ++kb) {
    const int keyBase = kb * 64;
    v8f sf[4];
#pragma unroll
    for (int sub = 0; sub < 4; ++sub) {
      v8f s = {};
      const int key = keyBase + sub * 16 + ln;
      const u16* kp = K + ((size_t)bh * Sc + key) * HDc + khB;
#pragma unroll
      for (int kc = 0; kc < 2; ++kc) {
        Frag kf;
        kf.f[0] = ldg4(kp + kc * 32);
        kf.f[1] = ldg4(kp + kc * 32 + 16);
        s = wmma_bf16(qa[kc].v, kf.v, s);
      }
      const int bark = bar[b * Sc + key];
      const int instk = inst[b * Sc + key];
      const bool kreal = instk < 129;
      const bool kglob = (instk == 129) || (bark == -1);
#pragma unroll
      for (int r = 0; r < 8; ++r) {
        const int qrow = qRowBase + r + 8 * half;
        bool vis;
        if (fc) {
          const int dd = bark - barq[r];
          const bool qreal = instq[r] < 129;
          const bool same = (instk == instq[r]) && kreal && qreal;
          const bool cross = (instk != instq[r]) && kreal && qreal &&
                             (dd == 0 || dd == 1 || dd == 2 || dd == 4);
          vis = (same && dd >= 0) || cross || kglob;
        } else {
          vis = true;
        }
        vis = vis && (key <= qrow);
        s[r] = vis ? s[r] * 0.125f : -3.0e38f;  // * HD^-0.5
      }
      sf[sub] = s;
    }
    // ---- online softmax (rows live per (r, lane-half); reduce 16 lanes) ----
    float alpha[8];
#pragma unroll
    for (int r = 0; r < 8; ++r) {
      float mx = fmaxf(fmaxf(sf[0][r], sf[1][r]), fmaxf(sf[2][r], sf[3][r]));
#pragma unroll
      for (int off = 1; off < 16; off <<= 1)
        mx = fmaxf(mx, __shfl_xor(mx, off, 32));
      const float mnew = fmaxf(mrow[r], mx);
      alpha[r] = __expf(mrow[r] - mnew);
      mrow[r] = mnew;
      float rs = 0.0f;
#pragma unroll
      for (int sub = 0; sub < 4; ++sub) {
        float p = (sf[sub][r] > -1.0e37f) ? __expf(sf[sub][r] - mnew) : 0.0f;
        sf[sub][r] = p;
        rs += p;
      }
#pragma unroll
      for (int off = 1; off < 16; off <<= 1) rs += __shfl_xor(rs, off, 32);
      lrow[r] = lrow[r] * alpha[r] + rs;
    }
#pragma unroll
    for (int nf = 0; nf < 4; ++nf)
#pragma unroll
      for (int r = 0; r < 8; ++r) o[nf][r] *= alpha[r];

    // ---- restage P (C-layout) through LDS into A-fragment layout ----
#pragma unroll
    for (int sub = 0; sub < 4; ++sub)
#pragma unroll
      for (int r = 0; r < 8; ++r)
        pl[(r + 8 * half) * 64 + sub * 16 + ln] = f2bf(sf[sub][r]);

    Frag pa[2];  // same-wave LDS ops are in-order (DScnt) -> no barrier needed
#pragma unroll
    for (int kc = 0; kc < 2; ++kc) {
      const u16* pp = pl + ln * 64 + kc * 32;
      pa[kc].f[0] = *(const v4f*)(pp + kh);
      pa[kc].f[1] = *(const v4f*)(pp + kh + 16);
    }

    // ---- O += P @ V  (K-dim = 64 keys, N = 64 HD cols) ----
#pragma unroll
    for (int nf = 0; nf < 4; ++nf) {
      const u16* vp = Vt + ((size_t)bh * HDc + nf * 16 + ln) * Sc + keyBase + khB;
#pragma unroll
      for (int kc = 0; kc < 2; ++kc) {
        Frag vf;
        vf.f[0] = ldg4(vp + kc * 32);
        vf.f[1] = ldg4(vp + kc * 32 + 16);
        o[nf] = wmma_bf16(pa[kc].v, vf.v, o[nf]);
      }
    }
  }

  // ---- finalize: O / l, store bf16 [B,S,D] for the Wo GEMM ----
#pragma unroll
  for (int r = 0; r < 8; ++r) {
    const float invl = (lrow[r] > 0.0f) ? 1.0f / lrow[r] : 0.0f;
    const int row = qRowBase + r + 8 * half;
#pragma unroll
    for (int nf = 0; nf < 4; ++nf) {
      O[(size_t)(b * Sc + row) * Dc + h * HDc + nf * 16 + ln] =
          f2bf(o[nf][r] * invl);
    }
  }
}

// ---------------------------------------------------------------------------
// Host launcher
// ---------------------------------------------------------------------------
extern "C" void kernel_launch(void* const* d_in, const int* in_sizes, int n_in,
                              void* d_out, int out_size, void* d_ws, size_t ws_size,
                              hipStream_t stream) {
  const float* x    = (const float*)d_in[0];
  const int*   bar  = (const int*)d_in[1];
  const int*   inst = (const int*)d_in[2];
  const int*   lyr  = (const int*)d_in[3];
  const float* Wq   = (const float*)d_in[4];
  const float* bq   = (const float*)d_in[5];
  const float* Wk   = (const float*)d_in[6];
  const float* bk   = (const float*)d_in[7];
  const float* Wv   = (const float*)d_in[8];
  const float* bv   = (const float*)d_in[9];
  const float* Wo   = (const float*)d_in[10];
  const float* bo   = (const float*)d_in[11];

  char* w = (char*)d_ws;
  size_t off = 0;
  auto alloc = [&](size_t bytes) {
    void* p = w + off;
    off = (off + bytes + 255) & ~(size_t)255;
    return p;
  };
  const size_t eMD = (size_t)Mrows * Dc;  // 4096*1024
  const size_t eDD = (size_t)Dc * Dc;     // 1024*1024
  u16* xb  = (u16*)alloc(eMD * 2);
  u16* Wqt = (u16*)alloc(eDD * 2);
  u16* Wkt = (u16*)alloc(eDD * 2);
  u16* Wvt = (u16*)alloc(eDD * 2);
  u16* Wot = (u16*)alloc(eDD * 2);
  u16* Qr  = (u16*)alloc(eMD * 2);  // [B,H,S,HD] bf16 (RoPE'd)
  u16* Kr  = (u16*)alloc(eMD * 2);  // [B,H,S,HD] bf16 (RoPE'd)
  u16* Vtp = (u16*)alloc(eMD * 2);  // [B,H,HD,S] bf16
  u16* Oat = (u16*)alloc(eMD * 2);  // [B,S,D]    bf16

  cvt_bf16_k<<<4096, 256, 0, stream>>>(x, xb, (int)eMD);
  cvt_wT_k<<<4096, 256, 0, stream>>>(Wq, Wqt);
  cvt_wT_k<<<4096, 256, 0, stream>>>(Wk, Wkt);
  cvt_wT_k<<<4096, 256, 0, stream>>>(Wv, Wvt);
  cvt_wT_k<<<4096, 256, 0, stream>>>(Wo, Wot);

  dim3 gGemm(Dc / 128, Mrows / 128);  // (8, 32)
  gemm_wmma_k<EPI_ROPE><<<gGemm, 256, 0, stream>>>(xb, Wqt, bq, Qr);
  gemm_wmma_k<EPI_ROPE><<<gGemm, 256, 0, stream>>>(xb, Wkt, bk, Kr);
  gemm_wmma_k<EPI_V><<<gGemm, 256, 0, stream>>>(xb, Wvt, bv, Vtp);

  attn_k<<<dim3(Bc * Hc, Sc / 64), 128, 0, stream>>>(Qr, Kr, Vtp, bar, inst,
                                                     lyr, Oat);

  gemm_wmma_k<EPI_OUT><<<gGemm, 256, 0, stream>>>(Oat, Wot, bo, d_out);
}
